// Attention_11038065951576
// MI455X (gfx1250) — matrix operand order repaired and checked by reading, not compiled, not run on hardware
//
#include <hip/hip_runtime.h>
#include <hip/hip_bf16.h>

// ---------------------------------------------------------------------------
// Multi-head attention (B=2, N=4096, C=512, H=8, D=64) for gfx1250 (MI455X).
// Flash-attention structure, bf16 WMMA (v_wmma_f32_16x16x32_bf16), f32 accum.
// Async global->LDS staging for K tiles (ASYNCcnt path).
// ---------------------------------------------------------------------------

#define BATCH    2
#define SEQ      4096
#define DIM      512
#define HEADS    8
#define HDIM     64
#define ROWS     (BATCH * SEQ)       // 8192
#define QKV_OUT  (3 * DIM)           // 1536
#define SCALE    0.125f              // 64^-0.5

typedef __attribute__((ext_vector_type(16))) __bf16 v16bf;
typedef __attribute__((ext_vector_type(8)))  __bf16 v8bf;
typedef __attribute__((ext_vector_type(8)))  float  v8f;
typedef __attribute__((ext_vector_type(4)))  float  v4f;
typedef __attribute__((ext_vector_type(4)))  int    v4i;

#if __has_builtin(__builtin_amdgcn_global_load_async_to_lds_b128)
#define HAVE_ASYNC_LDS 1
#else
#define HAVE_ASYNC_LDS 0
#endif

#define AS1 __attribute__((address_space(1)))
#define AS3 __attribute__((address_space(3)))

// float -> bf16: rely on native gfx1250 converts (v_cvt_*bf16*)
__device__ __forceinline__ __bf16 f2bf(float f) { return (__bf16)f; }

__device__ __forceinline__ v8f wmma_bf16(v16bf a, v16bf b, v8f c) {
    return __builtin_amdgcn_wmma_f32_16x16x32_bf16(
        /*neg_a=*/false, a, /*neg_b=*/false, b,
        /*c_mod=*/(short)0, c, /*reuse_a=*/false, /*reuse_b=*/false);
}

// A operand: 16x32 bf16 tile of row-major matrix `base` (leading dim ld),
// tile origin (row0, k0). Lane covers row (lane&15); elements 0..7 hold
// K = k0 + half*8 .. +7, elements 8..15 hold K = k0 + half*8 + 16 .. +23.
__device__ __forceinline__ v16bf load_afrag(const __bf16* base, int ld,
                                            int row0, int k0, int lane) {
    int l = lane & 15, half = lane >> 4;
    const __bf16* p = base + (size_t)(row0 + l) * ld + k0 + half * 8;
    v8bf lo = *reinterpret_cast<const v8bf*>(p);
    v8bf hi = *reinterpret_cast<const v8bf*>(p + 16);
    return __builtin_shufflevector(lo, hi, 0,1,2,3,4,5,6,7,8,9,10,11,12,13,14,15);
}

// B operand: 32x16 bf16 tile where conceptual B[k][n] is stored as
// base[n*ld + k] (transposed-weight layout). Lane covers column
// n0+(lane&15); elements 0..7 hold K = k0 + half*8 .. +7, elements 8..15 hold K = k0 + half*8 + 16 .. +23.
__device__ __forceinline__ v16bf load_bfrag(const __bf16* base, int ld,
                                            int n0, int k0, int lane) {
    int l = lane & 15, half = lane >> 4;
    const __bf16* p = base + (size_t)(n0 + l) * ld + k0 + half * 8;
    v8bf lo = *reinterpret_cast<const v8bf*>(p);
    v8bf hi = *reinterpret_cast<const v8bf*>(p + 16);
    return __builtin_shufflevector(lo, hi, 0,1,2,3,4,5,6,7,8,9,10,11,12,13,14,15);
}

// ---------------------------------------------------------------------------
// Kernel 0: f32 -> bf16 convert, 8 elements/thread (b128 in/out).
// All tensor sizes here are multiples of 8.
// ---------------------------------------------------------------------------
__global__ void cvt_bf16_kernel(const float* __restrict__ in,
                                __bf16* __restrict__ out, int n) {
    int i = (blockIdx.x * blockDim.x + threadIdx.x) * 8;
    if (i >= n) return;
    v4f a = *reinterpret_cast<const v4f*>(in + i);
    v4f b = *reinterpret_cast<const v4f*>(in + i + 4);
    v8bf r;
#pragma unroll
    for (int j = 0; j < 4; ++j) { r[j] = f2bf(a[j]); r[4 + j] = f2bf(b[j]); }
    *reinterpret_cast<v8bf*>(out + i) = r;
}

// ---------------------------------------------------------------------------
// Kernel 1: QKV projection.  Y[m,o] = sum_c X[m,c] * Wqkv[o,c]
// Each wave computes a 16x64 tile (A fragment reused across 4 WMMAs).
// Q results are pre-scaled by 1/sqrt(d).  Scatter into Q/K/V [B*H,SEQ,HDIM].
// Grid: (QKV_OUT/64, ROWS/128), block 256 (8 waves).
// ---------------------------------------------------------------------------
__global__ void qkv_gemm_kernel(const __bf16* __restrict__ xb,
                                const __bf16* __restrict__ wb,
                                __bf16* __restrict__ qb,
                                __bf16* __restrict__ kb,
                                __bf16* __restrict__ vb) {
    const int lane = threadIdx.x & 31;
    const int wave = threadIdx.x >> 5;
    const int m0 = blockIdx.y * 128 + wave * 16;
    const int n0 = blockIdx.x * 64;

    v8f acc[4];
#pragma unroll
    for (int j = 0; j < 4; ++j)
#pragma unroll
        for (int r = 0; r < 8; ++r) acc[j][r] = 0.f;

#pragma unroll 2
    for (int k0 = 0; k0 < DIM; k0 += 32) {
        v16bf a = load_afrag(xb, DIM, m0, k0, lane);
#pragma unroll
        for (int j = 0; j < 4; ++j) {
            v16bf b = load_bfrag(wb, DIM, n0 + 16 * j, k0, lane);
            acc[j] = wmma_bf16(a, b, acc[j]);
        }
    }

    const int l = lane & 15, half = lane >> 4;
#pragma unroll
    for (int j = 0; j < 4; ++j) {
        const int o = n0 + 16 * j + l;
        const int which = o >> 9;          // 0:q 1:k 2:v (uniform per subtile)
        const int oo = o & 511;
        const int h = oo >> 6;
        const int d = oo & 63;
        __bf16* dst = (which == 0) ? qb : (which == 1) ? kb : vb;
        const float sc = (which == 0) ? SCALE : 1.f;   // fold softmax scale
#pragma unroll
        for (int r = 0; r < 8; ++r) {
            int m = m0 + r + half * 8;
            int bidx = m >> 12;            // / SEQ
            int n = m & 4095;              // % SEQ
            dst[(((size_t)(bidx * HEADS + h) * SEQ) + n) * HDIM + d] =
                f2bf(acc[j][r] * sc);
        }
    }
}

// ---------------------------------------------------------------------------
// Kernel 2: flash attention.  One block = 4 waves = 64 query rows of one
// (batch,head).  64-key blocks: K staged to LDS asynchronously (ASYNCcnt),
// V transposed through registers so the PV B-operand loads contiguously.
// Online softmax in registers.  Grid: (B*H, SEQ/64), block 128.
// ---------------------------------------------------------------------------
__global__ void flash_attn_kernel(const __bf16* __restrict__ qb,
                                  const __bf16* __restrict__ kb,
                                  const __bf16* __restrict__ vb,
                                  __bf16* __restrict__ ob) {
    __shared__ __bf16 Kt[64 * 64];          // [key][c], row-major
    __shared__ __bf16 Vt[64 * 64];          // [d][key], transposed
    __shared__ __bf16 Pt[4][16 * 64];       // per-wave P tile, row-major

    const int tid  = threadIdx.x;
    const int lane = tid & 31;
    const int wave = tid >> 5;
    const int l = lane & 15, half = lane >> 4;

    const int bh = blockIdx.x;                    // 0..15
    const int b  = bh >> 3, h = bh & 7;
    const int q0 = blockIdx.y * 64 + wave * 16;   // this wave's query tile

    const __bf16* Qg = qb + (size_t)bh * SEQ * HDIM;
    const __bf16* Kg = kb + (size_t)bh * SEQ * HDIM;
    const __bf16* Vg = vb + (size_t)bh * SEQ * HDIM;

    // Q fragments (16x64 = two 16x32 A operands), resident all kernel.
    // (Q was pre-scaled by 1/sqrt(d) in the QKV kernel.)
    v16bf qf0 = load_afrag(Qg, HDIM, q0, 0, lane);
    v16bf qf1 = load_afrag(Qg, HDIM, q0, 32, lane);

    float mrun[8], lrun[8];
    v8f acc[4];
#pragma unroll
    for (int r = 0; r < 8; ++r) { mrun[r] = -1e30f; lrun[r] = 0.f; }
#pragma unroll
    for (int t = 0; t < 4; ++t)
#pragma unroll
        for (int j = 0; j < 8; ++j) acc[t][j] = 0.f;

    for (int kb0 = 0; kb0 < SEQ; kb0 += 64) {
        const __bf16* Kblk = Kg + (size_t)kb0 * HDIM;
        const __bf16* Vblk = Vg + (size_t)kb0 * HDIM;

        // ---- K tile: async global->LDS copy (layout preserved) ----
#pragma unroll
        for (int i = 0; i < 4; ++i) {
            int c   = tid + i * 128;      // 16B chunk id; 512 chunks total
            int key = c >> 3;
            int d8  = (c & 7) * 8;
#if HAVE_ASYNC_LDS
            __builtin_amdgcn_global_load_async_to_lds_b128(
                (AS1 v4i*)(Kblk + key * HDIM + d8),
                (AS3 v4i*)(Kt + key * 64 + d8), 0, 0);
#else
            *reinterpret_cast<v8bf*>(Kt + key * 64 + d8) =
                *reinterpret_cast<const v8bf*>(Kblk + key * HDIM + d8);
#endif
        }

        // ---- V tile: transpose through registers into LDS ----
#pragma unroll
        for (int i = 0; i < 4; ++i) {
            int c   = tid + i * 128;
            int key = c >> 3;
            int d8  = (c & 7) * 8;
            v8bf vv = *reinterpret_cast<const v8bf*>(Vblk + key * HDIM + d8);
#pragma unroll
            for (int j = 0; j < 8; ++j) Vt[(d8 + j) * 64 + key] = vv[j];
        }

#if HAVE_ASYNC_LDS
#if __has_builtin(__builtin_amdgcn_s_wait_asynccnt)
        __builtin_amdgcn_s_wait_asynccnt(0);
#else
        asm volatile("s_wait_asynccnt 0x0" ::: "memory");
#endif
#endif
        __syncthreads();

        // ---- S = Q * K^T  (4 sub-tiles of 16 keys) ----
        v8f s[4];
#pragma unroll
        for (int t = 0; t < 4; ++t) {
            v8f sv;
#pragma unroll
            for (int j = 0; j < 8; ++j) sv[j] = 0.f;
            sv = wmma_bf16(qf0, load_bfrag(Kt, 64, 16 * t, 0, lane), sv);
            sv = wmma_bf16(qf1, load_bfrag(Kt, 64, 16 * t, 32, lane), sv);
            s[t] = sv;
        }

        // ---- online softmax (row-wise over the 16-lane sub-group) ----
#pragma unroll
        for (int r = 0; r < 8; ++r) {
            float mx = fmaxf(fmaxf(s[0][r], s[1][r]), fmaxf(s[2][r], s[3][r]));
            mx = fmaxf(mx, __shfl_xor(mx, 1));
            mx = fmaxf(mx, __shfl_xor(mx, 2));
            mx = fmaxf(mx, __shfl_xor(mx, 4));
            mx = fmaxf(mx, __shfl_xor(mx, 8));
            float m_new = fmaxf(mrun[r], mx);
            float corr  = __expf(mrun[r] - m_new);
            float rs = 0.f;
#pragma unroll
            for (int t = 0; t < 4; ++t) {
                float p = __expf(s[t][r] - m_new);
                s[t][r] = p;
                rs += p;
            }
            rs += __shfl_xor(rs, 1);
            rs += __shfl_xor(rs, 2);
            rs += __shfl_xor(rs, 4);
            rs += __shfl_xor(rs, 8);
            lrun[r] = lrun[r] * corr + rs;
            mrun[r] = m_new;
#pragma unroll
            for (int t = 0; t < 4; ++t) acc[t][r] *= corr;
        }

        // ---- re-layout P: D-fragment -> LDS -> A-fragment ----
        __bf16* Pw = &Pt[wave][0];
#pragma unroll
        for (int t = 0; t < 4; ++t)
#pragma unroll
            for (int r = 0; r < 8; ++r)
                Pw[(r + half * 8) * 64 + 16 * t + l] = f2bf(s[t][r]);
        __syncthreads();

        // ---- O += P * V ----
#pragma unroll
        for (int ks = 0; ks < 2; ++ks) {
            v16bf pa = load_afrag(Pw, 64, 0, ks * 32, lane);
#pragma unroll
            for (int nt = 0; nt < 4; ++nt) {
                v16bf vbf = load_bfrag(Vt, 64, 16 * nt, ks * 32, lane);
                acc[nt] = wmma_bf16(pa, vbf, acc[nt]);
            }
        }
        __syncthreads();   // before next iteration overwrites Kt/Vt/Pt
    }

    // ---- normalize and write O as [B, N, H*D] bf16 ----
#pragma unroll
    for (int r = 0; r < 8; ++r) {
        float rinv = 1.f / lrun[r];
        int n = q0 + r + half * 8;
#pragma unroll
        for (int t = 0; t < 4; ++t) {
            ob[((size_t)b * SEQ + n) * DIM + h * HDIM + 16 * t + l] =
                f2bf(acc[t][r] * rinv);
        }
    }
}

// ---------------------------------------------------------------------------
// Kernel 3: output projection.  out[m,o] = sum_c O[m,c]*Wproj[o,c] + bias[o]
// Each wave computes a 16x64 tile.  Grid: (DIM/64, ROWS/128), block 256.
// ---------------------------------------------------------------------------
__global__ void proj_gemm_kernel(const __bf16* __restrict__ ob,
                                 const __bf16* __restrict__ wb,
                                 const float* __restrict__ bias,
                                 float* __restrict__ out) {
    const int lane = threadIdx.x & 31;
    const int wave = threadIdx.x >> 5;
    const int m0 = blockIdx.y * 128 + wave * 16;
    const int n0 = blockIdx.x * 64;

    v8f acc[4];
#pragma unroll
    for (int j = 0; j < 4; ++j)
#pragma unroll
        for (int r = 0; r < 8; ++r) acc[j][r] = 0.f;

#pragma unroll 2
    for (int k0 = 0; k0 < DIM; k0 += 32) {
        v16bf a = load_afrag(ob, DIM, m0, k0, lane);
#pragma unroll
        for (int j = 0; j < 4; ++j) {
            v16bf b = load_bfrag(wb, DIM, n0 + 16 * j, k0, lane);
            acc[j] = wmma_bf16(a, b, acc[j]);
        }
    }

    const int l = lane & 15, half = lane >> 4;
#pragma unroll
    for (int j = 0; j < 4; ++j) {
        const int o = n0 + 16 * j + l;
        const float bo = bias[o];
#pragma unroll
        for (int r = 0; r < 8; ++r) {
            int m = m0 + r + half * 8;
            out[(size_t)m * DIM + o] = acc[j][r] + bo;
        }
    }
}

// ---------------------------------------------------------------------------
// Launcher
// ---------------------------------------------------------------------------
extern "C" void kernel_launch(void* const* d_in, const int* in_sizes, int n_in,
                              void* d_out, int out_size, void* d_ws, size_t ws_size,
                              hipStream_t stream) {
    const float* x      = (const float*)d_in[0];   // [2,4096,512]
    const float* w_qkv  = (const float*)d_in[1];   // [1536,512]
    const float* w_proj = (const float*)d_in[2];   // [512,512]
    const float* b_proj = (const float*)d_in[3];   // [512]
    float* out = (float*)d_out;

    char* ws = (char*)d_ws;
    size_t off = 0;
    __bf16* XB = (__bf16*)(ws + off); off += (size_t)ROWS * DIM * 2;          // 8 MB
    __bf16* WQ = (__bf16*)(ws + off); off += (size_t)QKV_OUT * DIM * 2;       // 1.5 MB
    __bf16* WP = (__bf16*)(ws + off); off += (size_t)DIM * DIM * 2;           // 0.5 MB
    __bf16* Qb = (__bf16*)(ws + off); off += (size_t)BATCH * HEADS * SEQ * HDIM * 2;
    __bf16* Kb = (__bf16*)(ws + off); off += (size_t)BATCH * HEADS * SEQ * HDIM * 2;
    __bf16* Vb = (__bf16*)(ws + off); off += (size_t)BATCH * HEADS * SEQ * HDIM * 2;
    __bf16* Ob = (__bf16*)(ws + off); off += (size_t)ROWS * DIM * 2;          // 8 MB

    // 0) convert inputs to bf16 (vectorized, 8 elems/thread)
    {
        int n0 = ROWS * DIM;
        cvt_bf16_kernel<<<(n0 / 8 + 255) / 256, 256, 0, stream>>>(x, XB, n0);
        int n1 = QKV_OUT * DIM;
        cvt_bf16_kernel<<<(n1 / 8 + 255) / 256, 256, 0, stream>>>(w_qkv, WQ, n1);
        int n2 = DIM * DIM;
        cvt_bf16_kernel<<<(n2 / 8 + 255) / 256, 256, 0, stream>>>(w_proj, WP, n2);
    }

    // 1) QKV projection (Q pre-scaled by 1/sqrt(d))
    qkv_gemm_kernel<<<dim3(QKV_OUT / 64, ROWS / 128), 256, 0, stream>>>(
        XB, WQ, Qb, Kb, Vb);

    // 2) flash attention
    flash_attn_kernel<<<dim3(BATCH * HEADS, SEQ / 64), 128, 0, stream>>>(
        Qb, Kb, Vb, Ob);

    // 3) output projection + bias
    proj_gemm_kernel<<<dim3(DIM / 64, ROWS / 128), 256, 0, stream>>>(
        Ob, WP, b_proj, out);
}
